// MALAttention_58136677319430
// MI455X (gfx1250) — matrix operand order repaired and checked by reading, not compiled, not run on hardware
//
#include <hip/hip_runtime.h>
#include <cstdint>
#include <cstddef>

// ---------------------------------------------------------------------------
// MAL attention for MI455X (gfx1250, wave32, WMMA).
// Pipeline: f32->bf16 convert, QKV GEMM (WMMA bf16, double-buffered LDS with
// async global->LDS copies), q sigmoid-norm scale, flash attention (WMMA bf16,
// online softmax f32), proj GEMM + bias (f32 out).
// ---------------------------------------------------------------------------

typedef __attribute__((ext_vector_type(16))) __bf16   bf16x16;
typedef __attribute__((ext_vector_type(8)))  __bf16   bf16x8;
typedef __attribute__((ext_vector_type(4)))  __bf16   bf16x4;
typedef __attribute__((ext_vector_type(8)))  float    f32x8;
typedef __attribute__((ext_vector_type(4)))  float    f32x4;
typedef __attribute__((ext_vector_type(4)))  uint32_t u32x4;
typedef __attribute__((ext_vector_type(4)))  int      s32x4;

// Address-space qualified vector types for the async global->LDS builtins.
typedef __attribute__((address_space(1))) s32x4 gas_s32x4;
typedef __attribute__((address_space(3))) s32x4 las_s32x4;

union FragAB { bf16x16 v; u32x4 u[2]; };

static constexpr int   kBatch = 4;
static constexpr int   kSeq   = 2048;
static constexpr int   kDim   = 1024;
static constexpr int   kHeads = 16;
static constexpr int   kHD    = 64;
static constexpr int   kMAll  = kBatch * kSeq;               // 8192
static constexpr float kScale = 0.125f;                      // 64^-0.5
static constexpr size_t kPerG = (size_t)kBatch * kHeads * kSeq * kHD; // 8388608

__device__ __forceinline__ f32x8 wmma_bf16(const FragAB& a, const FragAB& b, f32x8 c) {
  return __builtin_amdgcn_wmma_f32_16x16x32_bf16(false, a.v, false, b.v,
                                                 (short)0, c, false, false);
}

// 16-byte global->LDS copy: async (ASYNCcnt-tracked) when available.
#if __has_builtin(__builtin_amdgcn_global_load_async_to_lds_b128)
#define MAL_ASYNC_COPY 1
#else
#define MAL_ASYNC_COPY 0
#endif

__device__ __forceinline__ void copy16(const __bf16* g, __bf16* l) {
#if MAL_ASYNC_COPY
  __builtin_amdgcn_global_load_async_to_lds_b128(
      (gas_s32x4*)(void*)g, (las_s32x4*)(void*)l, 0, 0);
#else
  *(u32x4*)l = *(const u32x4*)g;
#endif
}

__device__ __forceinline__ void wait_copies() {
#if MAL_ASYNC_COPY
#if __has_builtin(__builtin_amdgcn_s_wait_asynccnt)
  __builtin_amdgcn_s_wait_asynccnt(0);
#else
  asm volatile("s_wait_asynccnt 0" ::: "memory");
#endif
#endif
}

// ---------------------------------------------------------------------------
// f32 -> bf16 conversion, 4 elements per thread.
// ---------------------------------------------------------------------------
__global__ void mal_f32_to_bf16(const float* __restrict__ src,
                                __bf16* __restrict__ dst, int n4) {
  int i = blockIdx.x * blockDim.x + threadIdx.x;
  if (i >= n4) return;
  f32x4 f = *(const f32x4*)(src + 4 * (size_t)i);
  bf16x4 o;
  o.x = (__bf16)f.x; o.y = (__bf16)f.y; o.z = (__bf16)f.z; o.w = (__bf16)f.w;
  *(bf16x4*)(dst + 4 * (size_t)i) = o;
}

// ---------------------------------------------------------------------------
// Tiled bf16 WMMA GEMM:  C[M,N] = A[M,K] * W[N,K]^T
// 256 threads = 8 waves; 128x128 output tile; K in 64-wide panels,
// double-buffered LDS, async global->LDS staging overlapped with WMMA.
// Wave (wr,wc) owns a 32x64 sub-tile = 2x4 WMMA tiles.
// EPI==0: scatter bf16 into qkv buffer laid out [3][B][H][N][D].
// EPI==1: f32 output + bias (final projection into d_out).
// ---------------------------------------------------------------------------
template <int EPI>
__global__ __launch_bounds__(256)
void mal_gemm(const __bf16* __restrict__ A, const __bf16* __restrict__ W,
              int M, int N, int K,
              __bf16* __restrict__ qkv_out,
              float* __restrict__ out_f32, const float* __restrict__ bias) {
  __shared__ __attribute__((aligned(16))) __bf16 lA[2][128 * 64];
  __shared__ __attribute__((aligned(16))) __bf16 lB[2][128 * 64];

  const int t    = threadIdx.x;
  const int m0   = blockIdx.y * 128;
  const int n0   = blockIdx.x * 128;
  const int w    = t >> 5;
  const int lane = t & 31;
  const int lh   = lane & 15;
  const int sel  = lane >> 4;
  const int wr   = w >> 1;   // 0..3 -> 32-row band
  const int wc   = w & 1;    // 0..1 -> 64-col band

  const __bf16* Ab = A + (size_t)m0 * K;
  const __bf16* Wb = W + (size_t)n0 * K;

  auto stage = [&](int buf, int k0) {
#pragma unroll
    for (int c = t; c < 1024; c += 256) {
      int row = c >> 3, seg = c & 7;
      copy16(&Ab[(size_t)row * K + k0 + seg * 8], &lA[buf][row * 64 + seg * 8]);
      copy16(&Wb[(size_t)row * K + k0 + seg * 8], &lB[buf][row * 64 + seg * 8]);
    }
  };

  const f32x8 zero8 = {0.f, 0.f, 0.f, 0.f, 0.f, 0.f, 0.f, 0.f};
  f32x8 acc[2][4];
#pragma unroll
  for (int mt = 0; mt < 2; ++mt)
#pragma unroll
    for (int nt = 0; nt < 4; ++nt) acc[mt][nt] = zero8;

  stage(0, 0);
  wait_copies();
  __syncthreads();

  const int panels = K / 64;
  for (int p = 0; p < panels; ++p) {
    const int cur = p & 1;
    if (p + 1 < panels) stage(cur ^ 1, (p + 1) * 64);

    const char* lAc = (const char*)lA[cur];
    const char* lBc = (const char*)lB[cur];
#pragma unroll
    for (int kk = 0; kk < 64; kk += 32) {
      FragAB af[2], bf[4];
#pragma unroll
      for (int mt = 0; mt < 2; ++mt) {
        const char* base = lAc + (wr * 32 + mt * 16 + lh) * 128 + kk * 2 + sel * 16;
        af[mt].u[0] = *(const u32x4*)(base);
        af[mt].u[1] = *(const u32x4*)(base + 32);
      }
#pragma unroll
      for (int nt = 0; nt < 4; ++nt) {
        const char* base = lBc + (wc * 64 + nt * 16 + lh) * 128 + kk * 2 + sel * 16;
        bf[nt].u[0] = *(const u32x4*)(base);
        bf[nt].u[1] = *(const u32x4*)(base + 32);
      }
#pragma unroll
      for (int mt = 0; mt < 2; ++mt)
#pragma unroll
        for (int nt = 0; nt < 4; ++nt)
          acc[mt][nt] = wmma_bf16(af[mt], bf[nt], acc[mt][nt]);
    }
    wait_copies();
    __syncthreads();
  }

  // Epilogue. C/D layout: vgpr i, lane -> (row = i + 8*sel, col = lh) per tile.
#pragma unroll
  for (int mt = 0; mt < 2; ++mt) {
#pragma unroll
    for (int nt = 0; nt < 4; ++nt) {
#pragma unroll
      for (int i = 0; i < 8; ++i) {
        int m  = m0 + wr * 32 + mt * 16 + i + 8 * sel;
        int nn = n0 + wc * 64 + nt * 16 + lh;
        float val = acc[mt][nt][i];
        if (EPI == 0) {
          int g = nn >> 10, rem = nn & 1023;
          int h = rem >> 6, d = rem & 63;
          int b = m >> 11,  n = m & 2047;
          size_t idx = (size_t)g * kPerG +
                       (((size_t)(b * kHeads + h) * kSeq + n) * kHD + d);
          qkv_out[idx] = (__bf16)val;
        } else {
          out_f32[(size_t)m * N + nn] = val + bias[nn];
        }
      }
    }
  }
}

// ---------------------------------------------------------------------------
// q <- q * sigmoid(||q||) per (b,h,n) row of 64.
// ---------------------------------------------------------------------------
__global__ void mal_qscale(__bf16* __restrict__ q, int rows) {
  int r = blockIdx.x * blockDim.x + threadIdx.x;
  if (r >= rows) return;
  __bf16* p = q + (size_t)r * kHD;
  float vals[kHD];
  float s = 0.f;
#pragma unroll
  for (int i = 0; i < kHD; i += 8) {
    bf16x8 vv = *(const bf16x8*)(p + i);
#pragma unroll
    for (int j = 0; j < 8; ++j) {
      float f = (float)vv[j];
      vals[i + j] = f;
      s += f * f;
    }
  }
  float mag = sqrtf(s);
  float sg  = 1.f / (1.f + __expf(-mag));
#pragma unroll
  for (int i = 0; i < kHD; i += 8) {
    bf16x8 ov;
#pragma unroll
    for (int j = 0; j < 8; ++j) ov[j] = (__bf16)(vals[i + j] * sg);
    *(bf16x8*)(p + i) = ov;
  }
}

// ---------------------------------------------------------------------------
// Flash attention: one block per (b*h, 128-query tile). 8 waves, each wave
// owns 16 query rows. Stream 128-key panels; K staged row-major (async),
// V staged transposed (Vt[d][key]) so PV B-fragments are contiguous LDS
// reads. P round-trips through per-wave LDS to become an A fragment.
// ---------------------------------------------------------------------------
__global__ __launch_bounds__(256)
void mal_attn(const __bf16* __restrict__ Q, const __bf16* __restrict__ Kb,
              const __bf16* __restrict__ Vb, __bf16* __restrict__ Ob) {
  __shared__ __attribute__((aligned(16))) __bf16 lQ[128 * 64];
  __shared__ __attribute__((aligned(16))) __bf16 lK[128 * 64];
  __shared__ __attribute__((aligned(16))) __bf16 lVt[64 * 128];
  __shared__ __attribute__((aligned(16))) __bf16 lP[8 * 16 * 128];

  const int t    = threadIdx.x;
  const int bh   = blockIdx.x;   // b*16 + h
  const int qb   = blockIdx.y;   // query tile 0..15
  const int w    = t >> 5;
  const int lane = t & 31;
  const int lh   = lane & 15;
  const int sel  = lane >> 4;

  const __bf16* qp = Q + (size_t)bh * kSeq * kHD + (size_t)qb * 128 * kHD;
  const __bf16* kp = Kb + (size_t)bh * kSeq * kHD;
  const __bf16* vp = Vb + (size_t)bh * kSeq * kHD;

  // Stage Q tile (128x64) once (async copies).
#pragma unroll
  for (int c = t; c < 1024; c += 256) {
    int row = c >> 3, seg = c & 7;
    copy16(qp + row * 64 + seg * 8, &lQ[row * 64 + seg * 8]);
  }

  const f32x8 zero8 = {0.f, 0.f, 0.f, 0.f, 0.f, 0.f, 0.f, 0.f};
  f32x8 accO[4];
#pragma unroll
  for (int dt = 0; dt < 4; ++dt) accO[dt] = zero8;
  float mrow[8], lrow[8];
#pragma unroll
  for (int i = 0; i < 8; ++i) { mrow[i] = -3.0e38f; lrow[i] = 0.f; }

  for (int kb = 0; kb < kSeq / 128; ++kb) {
    const __bf16* kpp = kp + (size_t)kb * 128 * kHD;
    const __bf16* vpp = vp + (size_t)kb * 128 * kHD;
    // Prefetch next panel while this one is consumed.
    if (kb + 1 < kSeq / 128) {
      __builtin_prefetch(kpp + 128 * kHD + t * 32, 0, 1);
      __builtin_prefetch(vpp + 128 * kHD + t * 32, 0, 1);
    }
    __syncthreads();
#pragma unroll
    for (int c = t; c < 1024; c += 256) {
      int row = c >> 3, seg = c & 7;
      copy16(kpp + row * 64 + seg * 8, &lK[row * 64 + seg * 8]);
      bf16x8 vv = *(const bf16x8*)(vpp + row * 64 + seg * 8);
#pragma unroll
      for (int j = 0; j < 8; ++j) lVt[(seg * 8 + j) * 128 + row] = vv[j];
    }
    wait_copies();
    __syncthreads();

    // S = Q * K^T for this wave's 16 rows x 128 key cols.
    f32x8 accS[8];
#pragma unroll
    for (int ct = 0; ct < 8; ++ct) accS[ct] = zero8;
#pragma unroll
    for (int kk = 0; kk < kHD; kk += 32) {
      FragAB aq;
      const char* abase = (const char*)lQ + (w * 16 + lh) * 128 + kk * 2 + sel * 16;
      aq.u[0] = *(const u32x4*)abase;
      aq.u[1] = *(const u32x4*)(abase + 32);
#pragma unroll
      for (int ct = 0; ct < 8; ++ct) {
        FragAB bk;
        const char* bbase = (const char*)lK + (ct * 16 + lh) * 128 + kk * 2 + sel * 16;
        bk.u[0] = *(const u32x4*)bbase;
        bk.u[1] = *(const u32x4*)(bbase + 32);
        accS[ct] = wmma_bf16(aq, bk, accS[ct]);
      }
    }

    // Online softmax (f32 state, replicated across the 16 lanes of a half).
    __bf16* pw = lP + (size_t)w * 16 * 128;
#pragma unroll
    for (int i = 0; i < 8; ++i) {
      float mx = -3.0e38f;
#pragma unroll
      for (int ct = 0; ct < 8; ++ct) {
        accS[ct][i] *= kScale;
        mx = fmaxf(mx, accS[ct][i]);
      }
#pragma unroll
      for (int off = 8; off; off >>= 1) mx = fmaxf(mx, __shfl_xor(mx, off, 32));
      float mnew = fmaxf(mrow[i], mx);
      float corr = __expf(mrow[i] - mnew);
      lrow[i] *= corr;
#pragma unroll
      for (int dt = 0; dt < 4; ++dt) accO[dt][i] *= corr;
      float ps = 0.f;
      __bf16* prow = pw + (i + 8 * sel) * 128;
#pragma unroll
      for (int ct = 0; ct < 8; ++ct) {
        float p = __expf(accS[ct][i] - mnew);
        prow[ct * 16 + lh] = (__bf16)p;
        ps += p;
      }
#pragma unroll
      for (int off = 8; off; off >>= 1) ps += __shfl_xor(ps, off, 32);
      lrow[i] += ps;
      mrow[i] = mnew;
    }
    __syncthreads();

    // O += P * V  (K-dim = 128 keys, 4 WMMA steps).
#pragma unroll
    for (int ks = 0; ks < 4; ++ks) {
      FragAB ap;
      const char* pbase = (const char*)pw + lh * 256 + ks * 64 + sel * 16;
      ap.u[0] = *(const u32x4*)pbase;
      ap.u[1] = *(const u32x4*)(pbase + 32);
#pragma unroll
      for (int dt = 0; dt < 4; ++dt) {
        FragAB bv;
        const char* vbase =
            (const char*)lVt + (dt * 16 + lh) * 256 + ks * 64 + sel * 16;
        bv.u[0] = *(const u32x4*)vbase;
        bv.u[1] = *(const u32x4*)(vbase + 32);
        accO[dt] = wmma_bf16(ap, bv, accO[dt]);
      }
    }
  }

  // Normalize and scatter to [B, N, C] (C = h*64 + d) as bf16.
  const int b = bh >> 4, h = bh & 15;
#pragma unroll
  for (int i = 0; i < 8; ++i) {
    float inv = 1.f / lrow[i];
    int grow = b * kSeq + qb * 128 + w * 16 + i + 8 * sel;
#pragma unroll
    for (int dt = 0; dt < 4; ++dt) {
      int col = h * 64 + dt * 16 + lh;
      Ob[(size_t)grow * kDim + col] = (__bf16)(accO[dt][i] * inv);
    }
  }
}

// ---------------------------------------------------------------------------
// Launch.
// ---------------------------------------------------------------------------
extern "C" void kernel_launch(void* const* d_in, const int* in_sizes, int n_in,
                              void* d_out, int out_size, void* d_ws,
                              size_t ws_size, hipStream_t stream) {
  (void)in_sizes; (void)n_in; (void)out_size; (void)ws_size;
  const float* x      = (const float*)d_in[0];
  const float* w_qkv  = (const float*)d_in[1];
  const float* w_proj = (const float*)d_in[2];
  const float* b_proj = (const float*)d_in[3];
  float*       out    = (float*)d_out;

  char* ws = (char*)d_ws;
  __bf16* x_bf     = (__bf16*)(ws);                        // 16 MB
  __bf16* wqkv_bf  = (__bf16*)(ws + (16ull << 20));        //  6 MB
  __bf16* wproj_bf = (__bf16*)(ws + (22ull << 20));        //  2 MB
  __bf16* qkv_bf   = (__bf16*)(ws + (24ull << 20));        // 48 MB (q,k,v)
  __bf16* q_bf     = qkv_bf;
  __bf16* k_bf     = qkv_bf + kPerG;
  __bf16* v_bf     = qkv_bf + 2 * kPerG;
  __bf16* attn_bf  = (__bf16*)(ws + (72ull << 20));        // 16 MB

  const int nx = kMAll * kDim;          // 8388608
  const int nq = 3 * kDim * kDim;       // 3145728
  const int np = kDim * kDim;           // 1048576

  mal_f32_to_bf16<<<(nx / 4 + 255) / 256, 256, 0, stream>>>(x, x_bf, nx / 4);
  mal_f32_to_bf16<<<(nq / 4 + 255) / 256, 256, 0, stream>>>(w_qkv, wqkv_bf, nq / 4);
  mal_f32_to_bf16<<<(np / 4 + 255) / 256, 256, 0, stream>>>(w_proj, wproj_bf, np / 4);

  mal_gemm<0><<<dim3(3 * kDim / 128, kMAll / 128), 256, 0, stream>>>(
      x_bf, wqkv_bf, kMAll, 3 * kDim, kDim, qkv_bf, nullptr, nullptr);

  mal_qscale<<<(kBatch * kHeads * kSeq + 255) / 256, 256, 0, stream>>>(
      q_bf, kBatch * kHeads * kSeq);

  mal_attn<<<dim3(kBatch * kHeads, kSeq / 128), 256, 0, stream>>>(
      q_bf, k_bf, v_bf, attn_bf);

  mal_gemm<1><<<dim3(kDim / 128, kMAll / 128), 256, 0, stream>>>(
      attn_bf, wproj_bf, kMAll, kDim, kDim, nullptr, out, b_proj);
}
